// AttentionModel_41291815584395
// MI455X (gfx1250) — matrix operand order repaired and checked by reading, not compiled, not run on hardware
//
#include <hip/hip_runtime.h>
#include <hip/hip_bf16.h>

// ---------------------------------------------------------------------------
// Attention: O = softmax(mask(QK^T/sqrt(D))) @ V + V
// B=8, S=2048, D=1024.  All GEMMs via v_wmma_f32_16x16x32_f16 (wave32),
// software-pipelined (double-buffered K steps) so loads overlap WMMA.
// ---------------------------------------------------------------------------

#define BB 8
#define SS 2048
#define DD 1024

typedef _Float16 v16h __attribute__((ext_vector_type(16)));
typedef _Float16 v8h  __attribute__((ext_vector_type(8)));
typedef float    v8f  __attribute__((ext_vector_type(8)));
typedef float    v4f  __attribute__((ext_vector_type(4)));

__device__ __forceinline__ v16h cat16(v8h a, v8h b) {
  return __builtin_shufflevector(a, b, 0,1,2,3,4,5,6,7,8,9,10,11,12,13,14,15);
}

// ---- raw f32 operand fetch (QKV kernel) -----------------------------------
// A operand raw: lane holds row m: 8 floats at k0+h*8 and 8 at k0+16+h*8.
__device__ __forceinline__ void load_rawA(const float* __restrict__ X, int ld,
                                          int row0, int k0, int lane, v4f r[4]) {
  int m = lane & 15, h = lane >> 4;
  const float* p = X + (size_t)(row0 + m) * ld + k0 + h * 8;
  r[0] = *(const v4f*)(p);      r[1] = *(const v4f*)(p + 4);
  r[2] = *(const v4f*)(p + 16); r[3] = *(const v4f*)(p + 20);
}
// B operand raw: lane holds source row n: 8 floats at k0+h*8 and 8 at k0+16+h*8.
__device__ __forceinline__ void load_rawB(const float* __restrict__ W, int ld,
                                          int n0, int k0, int lane, v4f r[4]) {
  int n = lane & 15, h = lane >> 4;
  const float* p = W + (size_t)(n0 + n) * ld + k0 + h * 8;
  r[0] = *(const v4f*)(p);      r[1] = *(const v4f*)(p + 4);
  r[2] = *(const v4f*)(p + 16); r[3] = *(const v4f*)(p + 20);
}
__device__ __forceinline__ v16h cvt16(const v4f r[4]) {
  v16h o;
#pragma unroll
  for (int j = 0; j < 4; ++j)
#pragma unroll
    for (int i = 0; i < 4; ++i) o[j * 4 + i] = (_Float16)r[j][i];
  return o;
}

// ---- f16 operand fetch (scores / out kernels) -----------------------------
// A operand (16xK, row-major f16): lane<16: K=k0+0..7,k0+16..23 ; lane>=16:
// K=k0+8..15,k0+24..31   (ISA 16-bit A layout)
__device__ __forceinline__ v16h load_a(const _Float16* __restrict__ src, int ld,
                                       int row0, int k0, int lane) {
  int m = lane & 15, h = lane >> 4;
  const _Float16* r = src + (size_t)(row0 + m) * ld + k0 + h * 8;
  return cat16(*(const v8h*)(r), *(const v8h*)(r + 16));
}
// B operand (Kx16): lane n holds source row n, k at k0+h*8..+7 and k0+16+h*8..+7.
__device__ __forceinline__ v16h load_b(const _Float16* __restrict__ src, int ld,
                                       int n0, int k0, int lane) {
  int n = lane & 15, h = lane >> 4;
  const _Float16* r = src + (size_t)(n0 + n) * ld + k0 + h * 8;
  return cat16(*(const v8h*)(r), *(const v8h*)(r + 16));
}

__device__ __forceinline__ v8f wmma(v16h a, v16h b, v8f c) {
  return __builtin_amdgcn_wmma_f32_16x16x32_f16(false, a, false, b, (short)0, c,
                                                false, false);
}

// ---------------------------------------------------------------------------
// K1: Q/K/V projections.  out[r,n] = sum_d X[r,d]*W[n,d] + b[n]
// grid: x = (B*S)/16 row tiles, y*4waves = D/64 col groups, z = {Q,K,V}
// Q,K row-major f16 [B*S,D]; V transposed f16 [B,D,S].
// Pipelined: next-k raw f32 loads issue before current WMMAs; cvt after.
// ---------------------------------------------------------------------------
__global__ __launch_bounds__(128) void qkv_kernel(
    const float* __restrict__ x1, const float* __restrict__ x2,
    const float* __restrict__ x3, const float* __restrict__ wq,
    const float* __restrict__ bq, const float* __restrict__ wk,
    const float* __restrict__ bk, const float* __restrict__ wv,
    const float* __restrict__ bv, _Float16* __restrict__ Q,
    _Float16* __restrict__ K, _Float16* __restrict__ Vt) {
  const int z = blockIdx.z;
  const float* X  = (z == 0) ? x1 : (z == 1) ? x2 : x3;
  const float* W  = (z == 0) ? wq : (z == 1) ? wk : wv;
  const float* Bv = (z == 0) ? bq : (z == 1) ? bk : bv;

  const int lane = threadIdx.x & 31;
  const int wave = threadIdx.x >> 5;
  const int row0 = blockIdx.x * 16;
  const int n0   = (blockIdx.y * 4 + wave) * 64;

  v8f acc[4] = {};
  v4f ra[4], rb[4][4];
  load_rawA(X, DD, row0, 0, lane, ra);
#pragma unroll
  for (int t = 0; t < 4; ++t) load_rawB(W, DD, n0 + t * 16, 0, lane, rb[t]);
  v16h a = cvt16(ra);
  v16h bh[4];
#pragma unroll
  for (int t = 0; t < 4; ++t) bh[t] = cvt16(rb[t]);

  for (int k0 = 32; k0 < DD; k0 += 32) {
    // (1) prefetch next-k raw operands (fresh registers, no wait yet)
    load_rawA(X, DD, row0, k0, lane, ra);
#pragma unroll
    for (int t = 0; t < 4; ++t) load_rawB(W, DD, n0 + t * 16, k0, lane, rb[t]);
    // (2) matrix ops on current operands
#pragma unroll
    for (int t = 0; t < 4; ++t) acc[t] = wmma(a, bh[t], acc[t]);
    // (3) wait + convert next -> current (overlaps next trip's loads)
    a = cvt16(ra);
#pragma unroll
    for (int t = 0; t < 4; ++t) bh[t] = cvt16(rb[t]);
  }
#pragma unroll
  for (int t = 0; t < 4; ++t) acc[t] = wmma(a, bh[t], acc[t]);

  const int m = lane & 15, h = lane >> 4;
#pragma unroll
  for (int t = 0; t < 4; ++t) {
    int n = n0 + t * 16 + m;
    float bias = Bv[n];
#pragma unroll
    for (int i = 0; i < 8; ++i) {
      int r = row0 + i + h * 8;
      _Float16 hv = (_Float16)(acc[t][i] + bias);
      if (z == 0) {
        Q[(size_t)r * DD + n] = hv;
      } else if (z == 1) {
        K[(size_t)r * DD + n] = hv;
      } else {
        int bidx = r >> 11, s = r & (SS - 1);
        Vt[((size_t)bidx * DD + n) * SS + s] = hv;
      }
    }
  }
}

// ---------------------------------------------------------------------------
// K2: scores[b,q,k] = (Q[b,q,:].K[b,k,:])/32, -inf for k >= seqlen[b].
// grid: x = S/16 qtiles, y*4waves = S/64 kgroups, z = B.  Pipelined.
// ---------------------------------------------------------------------------
__global__ __launch_bounds__(128) void scores_kernel(
    const _Float16* __restrict__ Q, const _Float16* __restrict__ K,
    const int* __restrict__ seqlen, _Float16* __restrict__ Sc) {
  const int lane = threadIdx.x & 31;
  const int wave = threadIdx.x >> 5;
  const int b = blockIdx.z;
  const int q0 = blockIdx.x * 16;
  const int kc0 = (blockIdx.y * 4 + wave) * 64;
  const _Float16* Qb = Q + (size_t)b * SS * DD;
  const _Float16* Kb = K + (size_t)b * SS * DD;

  v8f acc[4] = {};
  v16h a = load_a(Qb, DD, q0, 0, lane);
  v16h bh[4];
#pragma unroll
  for (int t = 0; t < 4; ++t) bh[t] = load_b(Kb, DD, kc0 + t * 16, 0, lane);

  for (int d0 = 32; d0 < DD; d0 += 32) {
    v16h an = load_a(Qb, DD, q0, d0, lane);
    v16h bn[4];
#pragma unroll
    for (int t = 0; t < 4; ++t) bn[t] = load_b(Kb, DD, kc0 + t * 16, d0, lane);
#pragma unroll
    for (int t = 0; t < 4; ++t) acc[t] = wmma(a, bh[t], acc[t]);
    a = an;
#pragma unroll
    for (int t = 0; t < 4; ++t) bh[t] = bn[t];
  }
#pragma unroll
  for (int t = 0; t < 4; ++t) acc[t] = wmma(a, bh[t], acc[t]);

  const int sl = seqlen[b];
  const int m = lane & 15, h = lane >> 4;
  const float scale = 0.03125f;  // 1/sqrt(1024)
  const float ninf = -__builtin_inff();
#pragma unroll
  for (int t = 0; t < 4; ++t) {
    int kk = kc0 + t * 16 + m;
    bool masked = (kk >= sl);
#pragma unroll
    for (int i = 0; i < 8; ++i) {
      float v = masked ? ninf : acc[t][i] * scale;
      Sc[((size_t)b * SS + q0 + i + h * 8) * SS + kk] = (_Float16)v;
    }
  }
}

// ---------------------------------------------------------------------------
// K3: in-place row softmax over S=2048 columns; one 256-thread block per row.
// ---------------------------------------------------------------------------
__global__ __launch_bounds__(256) void softmax_kernel(_Float16* __restrict__ Sc) {
  __shared__ float red[256];
  const int t = threadIdx.x;
  _Float16* row = Sc + (size_t)blockIdx.x * SS;

  float vals[8];
  float mx = -__builtin_inff();
#pragma unroll
  for (int i = 0; i < 8; ++i) {
    vals[i] = (float)row[t + i * 256];
    mx = fmaxf(mx, vals[i]);
  }
  red[t] = mx; __syncthreads();
  for (int off = 128; off > 0; off >>= 1) {
    if (t < off) red[t] = fmaxf(red[t], red[t + off]);
    __syncthreads();
  }
  mx = red[0]; __syncthreads();

  float sum = 0.f;
#pragma unroll
  for (int i = 0; i < 8; ++i) {
    vals[i] = __expf(vals[i] - mx);   // masked (-inf) -> 0
    sum += vals[i];
  }
  red[t] = sum; __syncthreads();
  for (int off = 128; off > 0; off >>= 1) {
    if (t < off) red[t] += red[t + off];
    __syncthreads();
  }
  float inv = 1.0f / red[0];
#pragma unroll
  for (int i = 0; i < 8; ++i) row[t + i * 256] = (_Float16)(vals[i] * inv);
}

// ---------------------------------------------------------------------------
// K4: out[b,q,d] = sum_k P[b,q,k]*Vt[b,d,k] + Vt[b,d,q]   (f32 out). Pipelined.
// grid: x = S/16 qtiles, y*4waves = D/64 dgroups, z = B
// ---------------------------------------------------------------------------
__global__ __launch_bounds__(128) void out_kernel(
    const _Float16* __restrict__ P, const _Float16* __restrict__ Vt,
    float* __restrict__ out) {
  const int lane = threadIdx.x & 31;
  const int wave = threadIdx.x >> 5;
  const int b = blockIdx.z;
  const int q0 = blockIdx.x * 16;
  const int d0 = (blockIdx.y * 4 + wave) * 64;
  const _Float16* Pb = P + (size_t)b * SS * SS;
  const _Float16* Vb = Vt + (size_t)b * DD * SS;

  v8f acc[4] = {};
  v16h a = load_a(Pb, SS, q0, 0, lane);
  v16h bh[4];
#pragma unroll
  for (int t = 0; t < 4; ++t) bh[t] = load_b(Vb, SS, d0 + t * 16, 0, lane);

  for (int k0 = 32; k0 < SS; k0 += 32) {
    v16h an = load_a(Pb, SS, q0, k0, lane);
    v16h bn[4];
#pragma unroll
    for (int t = 0; t < 4; ++t) bn[t] = load_b(Vb, SS, d0 + t * 16, k0, lane);
#pragma unroll
    for (int t = 0; t < 4; ++t) acc[t] = wmma(a, bh[t], acc[t]);
    a = an;
#pragma unroll
    for (int t = 0; t < 4; ++t) bh[t] = bn[t];
  }
#pragma unroll
  for (int t = 0; t < 4; ++t) acc[t] = wmma(a, bh[t], acc[t]);

  const int m = lane & 15, h = lane >> 4;
#pragma unroll
  for (int t = 0; t < 4; ++t) {
    int d = d0 + t * 16 + m;
#pragma unroll
    for (int i = 0; i < 8; ++i) {
      int q = q0 + i + h * 8;
      float v = acc[t][i] + (float)Vb[(size_t)d * SS + q];  // + V[b,q,d]
      out[((size_t)b * SS + q) * DD + d] = v;
    }
  }
}

// ---------------------------------------------------------------------------
extern "C" void kernel_launch(void* const* d_in, const int* in_sizes, int n_in,
                              void* d_out, int out_size, void* d_ws,
                              size_t ws_size, hipStream_t stream) {
  const float* plms1 = (const float*)d_in[0];
  const float* plms2 = (const float*)d_in[1];
  const float* plms3 = (const float*)d_in[2];
  const int*   seql  = (const int*)d_in[3];
  const float* Wq = (const float*)d_in[4];
  const float* bq = (const float*)d_in[5];
  const float* Wk = (const float*)d_in[6];
  const float* bk = (const float*)d_in[7];
  const float* Wv = (const float*)d_in[8];
  const float* bv = (const float*)d_in[9];

  const size_t nQKV = (size_t)BB * SS * DD;
  _Float16* Q  = (_Float16*)d_ws;
  _Float16* K  = Q + nQKV;
  _Float16* Vt = K + nQKV;                    // [B, D, S]
  _Float16* Sc = Vt + nQKV;                   // [B, S, S]

  qkv_kernel<<<dim3((BB * SS) / 16, DD / 64 / 4, 3), 128, 0, stream>>>(
      plms1, plms2, plms3, Wq, bq, Wk, bk, Wv, bv, Q, K, Vt);
  scores_kernel<<<dim3(SS / 16, SS / 64 / 4, BB), 128, 0, stream>>>(Q, K, seql, Sc);
  softmax_kernel<<<dim3(BB * SS), 256, 0, stream>>>(Sc);
  out_kernel<<<dim3(SS / 16, DD / 64 / 4, BB), 128, 0, stream>>>(Sc, Vt,
                                                                 (float*)d_out);
}
